// EncoderLayer_60765197304090
// MI455X (gfx1250) — compile-verified, matrix operand order checked by reading, not run on hardware
//
#include <hip/hip_runtime.h>

// ---------------- problem constants ----------------
constexpr int NB  = 4;
constexpr int NS  = 2048;
constexpr int ND  = 1024;
constexpr int NH  = 16;
constexpr int NDK = 64;
constexpr int NFF = 4096;

typedef _Float16 v16h __attribute__((ext_vector_type(16)));
typedef _Float16 v8h  __attribute__((ext_vector_type(8)));
typedef _Float16 v4hh __attribute__((ext_vector_type(4)));
typedef float    v8f  __attribute__((ext_vector_type(8)));

__device__ __forceinline__ v8f wmma_f16(v16h a, v16h b, v8f c) {
  // 8 args: (neg_a, A, neg_b, B, c_mod, C, reuse_a, reuse_b)
  return __builtin_amdgcn_wmma_f32_16x16x32_f16(false, a, false, b, (short)0, c,
                                                false, false);
}

// Async DMA: copy 16B per lane from global memory directly into LDS.
// Tracked by ASYNCcnt; completion enforced with s_wait_asynccnt.
__device__ __forceinline__ void async_b128(const _Float16* src, unsigned lds_off) {
  asm volatile("global_load_async_to_lds_b128 %0, %1, off"
               :: "v"(lds_off), "v"(src) : "memory");
}

// Load one 16-half WMMA A/B fragment covering K=0..31 from an LDS row whose
// K axis is contiguous. Halves 0..7 -> K = 8*hi+0..7, halves 8..15 ->
// K = 16+8*hi+0..7: two 16B-aligned groups -> two ds_load_b128.
__device__ __forceinline__ v16h frag_ld(const _Float16* row, int hi) {
  const v8h lo = *(const v8h*)(row + hi * 8);
  const v8h hh = *(const v8h*)(row + 16 + hi * 8);
  return __builtin_shufflevector(lo, hh, 0, 1, 2, 3, 4, 5, 6, 7,
                                 8, 9, 10, 11, 12, 13, 14, 15);
}

// ---------------- pre-pass: f32 -> f16 convert / convert+transpose ----------------
__global__ __launch_bounds__(256) void cvt_kernel(const float* __restrict__ s,
                                                  _Float16* __restrict__ d) {
  const long i = ((long)blockIdx.x * 256 + threadIdx.x) * 4;
  const float4 f = *(const float4*)(s + i);
  d[i + 0] = (_Float16)f.x; d[i + 1] = (_Float16)f.y;
  d[i + 2] = (_Float16)f.z; d[i + 3] = (_Float16)f.w;
}

// W [K,N] f32 -> WT [N,K] f16 (so GEMM B-tiles are contiguous along K)
__global__ __launch_bounds__(256) void transpose_cvt_kernel(
    const float* __restrict__ W, _Float16* __restrict__ WT, int K, int N) {
  __shared__ float t[32][33];
  const int tx = threadIdx.x & 31;
  const int ty = threadIdx.x >> 5;  // 0..7
  const int kb = blockIdx.y * 32;
  const int nb = blockIdx.x * 32;
#pragma unroll
  for (int i = 0; i < 4; ++i)
    t[ty + i * 8][tx] = W[(long)(kb + ty + i * 8) * N + nb + tx];
  __syncthreads();
#pragma unroll
  for (int i = 0; i < 4; ++i)
    WT[(long)(nb + ty + i * 8) * K + kb + tx] = (_Float16)t[tx][ty + i * 8];
}

// ------- f16 WMMA GEMM, async double-buffered tiles: Y = act(A @ Bt^T + bias) -------
constexpr int BM = 128, BN = 128, BK = 32;
constexpr int TS_STR = 48;  // halves per tile row (32 + pad; 96B keeps 16B align)

template <int MODE>  // 0 = none, 1 = relu
__global__ __launch_bounds__(256) void gemm16_kernel(
    const _Float16* __restrict__ Ah, const _Float16* __restrict__ Bt,
    const float* __restrict__ bias, _Float16* __restrict__ Y,
    int M, int N, int K) {
  __shared__ _Float16 As[2][BM * TS_STR];  // [m][k], double-buffered
  __shared__ _Float16 Bs[2][BN * TS_STR];  // [n][k], double-buffered

  const int tid   = threadIdx.x;
  const int lane  = tid & 31;
  const int wid   = tid >> 5;     // 8 waves
  const int waveM = wid >> 2;     // 0..1 : 64 rows each
  const int waveN = wid & 3;      // 0..3 : 32 cols each
  const int hi    = lane >> 4;
  const int l15   = lane & 15;
  const long bm   = (long)blockIdx.y * BM;
  const long bn   = (long)blockIdx.x * BN;

  v8f zero = {};
  v8f acc[4][2];
#pragma unroll
  for (int i = 0; i < 4; ++i)
#pragma unroll
    for (int j = 0; j < 2; ++j) acc[i][j] = zero;

  // 128x32-half tile = 512 x 16B chunks; 256 lanes x 2 chunks each;
  // 4 async instructions per wave per (A,B) tile pair.
  auto issue = [&](int kb, int p) {
#pragma unroll
    for (int i = 0; i < 2; ++i) {
      const int c = wid * 64 + i * 32 + lane;  // 0..511
      const int row = c >> 2, qt = c & 3;
      async_b128(Ah + (bm + row) * (long)K + kb + qt * 8,
                 (unsigned)(size_t)&As[p][row * TS_STR + qt * 8]);
    }
#pragma unroll
    for (int i = 0; i < 2; ++i) {
      const int c = wid * 64 + i * 32 + lane;
      const int row = c >> 2, qt = c & 3;
      async_b128(Bt + (bn + row) * (long)K + kb + qt * 8,
                 (unsigned)(size_t)&Bs[p][row * TS_STR + qt * 8]);
    }
  };

  issue(0, 0);
  const int T = K / BK;
  for (int t = 0; t < T; ++t) {
    const int p = t & 1;
    if (t + 1 < T) {
      issue((t + 1) * BK, p ^ 1);
      // 4 ops of tile t+1 outstanding; ASYNCcnt loads complete in order,
      // so <=4 guarantees tile t fully landed in LDS.
      asm volatile("s_wait_asynccnt 0x4" ::: "memory");
    } else {
      asm volatile("s_wait_asynccnt 0x0" ::: "memory");
    }
    __syncthreads();

    v16h af[4], bf[2];
#pragma unroll
    for (int tm = 0; tm < 4; ++tm) {
      const int m = waveM * 64 + tm * 16 + l15;
      af[tm] = frag_ld(&As[p][m * TS_STR], hi);
    }
#pragma unroll
    for (int tn = 0; tn < 2; ++tn) {
      const int n = waveN * 32 + tn * 16 + l15;
      bf[tn] = frag_ld(&Bs[p][n * TS_STR], hi);
    }
#pragma unroll
    for (int tm = 0; tm < 4; ++tm)
#pragma unroll
      for (int tn = 0; tn < 2; ++tn)
        acc[tm][tn] = wmma_f16(af[tm], bf[tn], acc[tm][tn]);
    __syncthreads();
  }

  // epilogue: D-layout row M = r + 8*hi, col N = lane&15
#pragma unroll
  for (int tm = 0; tm < 4; ++tm)
#pragma unroll
    for (int tn = 0; tn < 2; ++tn) {
      const long gn = bn + waveN * 32 + tn * 16 + l15;
      const float bv = bias[gn];
#pragma unroll
      for (int r = 0; r < 8; ++r) {
        const long gm = bm + waveM * 64 + tm * 16 + r + hi * 8;
        float y = acc[tm][tn][r] + bv;
        if (MODE == 1) y = fmaxf(y, 0.0f);
        Y[gm * (long)N + gn] = (_Float16)y;
      }
    }
}

// ---------------- flash-attention (f16 Q/K/V, one 16-query tile per wave) ----------------
constexpr int KL_STR = 72;  // [key][d] row stride (144B, 16B-aligned)
constexpr int VT_STR = 48;  // [d][key]
constexpr int PP_STR = 48;  // [q][key]

__global__ __launch_bounds__(128) void attn_kernel(
    const _Float16* __restrict__ Qh, const _Float16* __restrict__ Kh,
    const _Float16* __restrict__ Vh, const int* __restrict__ mask,
    _Float16* __restrict__ O) {
  __shared__ _Float16 kl[4][32][KL_STR];   // K chunk  [key][d] (async-filled)
  __shared__ _Float16 vlT[4][64][VT_STR];  // V chunk  [d][key] (transposed)
  __shared__ _Float16 pP[4][16][PP_STR];   // P tile   [q][key]

  const int tid   = threadIdx.x;
  const int w     = tid >> 5;
  const int lane  = tid & 31;
  const int hi    = lane >> 4;
  const int qn    = lane & 15;
  const int b     = blockIdx.z;
  const int head  = blockIdx.y;
  const int qbase = blockIdx.x * 64 + w * 16;
  const float scale = 0.125f;  // 1/sqrt(64)

  // Q^T fragments (B operand: N = query, K = head dim), d 0..31 and 32..63
  v16h qf0, qf1;
  {
    const long qoff = ((long)(b * NS + qbase + qn) * NH + head) * NDK;
#pragma unroll
    for (int e = 0; e < 16; ++e) {
      const int d = (e & 7) + ((e >> 3) << 4) + hi * 8;
      qf0[e] = Qh[qoff + d];
      qf1[e] = Qh[qoff + 32 + d];
    }
  }

  float m_run = -3.0e38f, l_run = 0.0f;
  v8f zero = {};
  v8f acc[4] = {zero, zero, zero, zero};

  const int vb_k = (lane & 7) * 4;   // V transpose block origin (key)
  const int vb_d = (lane >> 3) * 4;  // V transpose block origin (d), +16*bi

  for (int kb = 0; kb < NS; kb += 32) {
    // ---- async-stage K chunk (32 keys x 64 d = 4KB f16): 8 b128 copies/wave ----
    {
      const long roff = ((long)(b * NS + kb) * NH + head) * NDK;
#pragma unroll
      for (int i = 0; i < 8; ++i) {
        const int c = i * 32 + lane;  // 0..255
        const int row = c >> 3, qt = c & 7;
        async_b128(Kh + roff + (long)row * NH * NDK + qt * 8,
                   (unsigned)(size_t)&kl[w][row][qt * 8]);
      }
    }
    // ---- stage V chunk with 4x4 register transpose -> [d][key] (overlaps async) ----
#pragma unroll
    for (int bi = 0; bi < 4; ++bi) {
      const int d0 = vb_d + bi * 16;
      const _Float16* s =
          Vh + ((long)(b * NS + kb + vb_k) * NH + head) * NDK + d0;
      const long rs = (long)NH * NDK;
      const v4hh r0 = *(const v4hh*)(s + 0 * rs);
      const v4hh r1 = *(const v4hh*)(s + 1 * rs);
      const v4hh r2 = *(const v4hh*)(s + 2 * rs);
      const v4hh r3 = *(const v4hh*)(s + 3 * rs);
      _Float16* t0 = &vlT[w][d0 + 0][vb_k];
      _Float16* t1 = &vlT[w][d0 + 1][vb_k];
      _Float16* t2 = &vlT[w][d0 + 2][vb_k];
      _Float16* t3 = &vlT[w][d0 + 3][vb_k];
      t0[0]=r0[0]; t0[1]=r1[0]; t0[2]=r2[0]; t0[3]=r3[0];
      t1[0]=r0[1]; t1[1]=r1[1]; t1[2]=r2[1]; t1[3]=r3[1];
      t2[0]=r0[2]; t2[1]=r1[2]; t2[2]=r2[2]; t2[3]=r3[2];
      t3[0]=r0[3]; t3[1]=r1[3]; t3[2]=r2[3]; t3[3]=r3[3];
    }
    asm volatile("s_wait_asynccnt 0x0" ::: "memory");

    // ---- S^T = K_tile (16keys x 64) @ Q^T (64 x 16q), two 16-key subtiles ----
    float st[2][8];
#pragma unroll
    for (int t = 0; t < 2; ++t) {
      const v16h kf0 = frag_ld(&kl[w][t * 16 + qn][0], hi);
      const v16h kf1 = frag_ld(&kl[w][t * 16 + qn][32], hi);
      v8f s = zero;
      s = wmma_f16(kf0, qf0, s);
      s = wmma_f16(kf1, qf1, s);
#pragma unroll
      for (int r = 0; r < 8; ++r) {
        const int key = kb + t * 16 + r + hi * 8;
        st[t][r] = (mask[b * NS + key] != 0) ? s[r] * scale : -1.0e9f;
      }
    }

    // ---- online softmax (per-query stats live at lanes with qn == q) ----
    float cm = -3.0e38f;
#pragma unroll
    for (int t = 0; t < 2; ++t)
#pragma unroll
      for (int r = 0; r < 8; ++r) cm = fmaxf(cm, st[t][r]);
    cm = fmaxf(cm, __shfl_xor(cm, 16, 32));
    const float m_new = fmaxf(m_run, cm);
    const float alpha = __expf(m_run - m_new);
    float rs = 0.0f;
#pragma unroll
    for (int t = 0; t < 2; ++t)
#pragma unroll
      for (int r = 0; r < 8; ++r) {
        const float p = __expf(st[t][r] - m_new);
        pP[w][qn][t * 16 + r + hi * 8] = (_Float16)p;  // transpose via LDS
        rs += p;
      }
    rs += __shfl_xor(rs, 16, 32);
    l_run = l_run * alpha + rs;
    m_run = m_new;

    // rescale running ctx accumulators (row q = r + 8*hi)
    float arow[8];
#pragma unroll
    for (int r = 0; r < 8; ++r) arow[r] = __shfl(alpha, r + hi * 8, 32);
#pragma unroll
    for (int j = 0; j < 4; ++j)
#pragma unroll
      for (int r = 0; r < 8; ++r) acc[j][r] *= arow[r];

    // ---- ctx += P (16q x 32keys) @ V (32keys x 64d) ----
    const v16h pf = frag_ld(&pP[w][qn][0], hi);
#pragma unroll
    for (int j = 0; j < 4; ++j) {
      const v16h vf = frag_ld(&vlT[w][j * 16 + qn][0], hi);
      acc[j] = wmma_f16(pf, vf, acc[j]);
    }
  }

  float linv[8];
#pragma unroll
  for (int r = 0; r < 8; ++r) linv[r] = 1.0f / __shfl(l_run, r + hi * 8, 32);
#pragma unroll
  for (int j = 0; j < 4; ++j)
#pragma unroll
    for (int r = 0; r < 8; ++r) {
      const long o =
          ((long)(b * NS + qbase + r + hi * 8) * NH + head) * NDK + j * 16 + qn;
      O[o] = (_Float16)(acc[j][r] * linv[r]);
    }
}

// ---------------- residual + LayerNorm (unbiased std, /(std+eps)) ----------------
template <typename OT>
__global__ __launch_bounds__(256) void add_ln_kernel(
    const _Float16* __restrict__ X, const _Float16* __restrict__ R,
    const float* __restrict__ gamma, const float* __restrict__ beta,
    OT* __restrict__ out) {
  __shared__ float red[8];
  __shared__ float s_mean, s_inv;
  const int tid  = threadIdx.x;
  const int lane = tid & 31, wid = tid >> 5;
  const long row = blockIdx.x;
  const _Float16* xp = X + row * ND;
  const _Float16* rp = R + row * ND;
  float v[4];
#pragma unroll
  for (int i = 0; i < 4; ++i)
    v[i] = (float)xp[tid + i * 256] + (float)rp[tid + i * 256];
  float s = v[0] + v[1] + v[2] + v[3];
#pragma unroll
  for (int o = 16; o >= 1; o >>= 1) s += __shfl_xor(s, o, 32);
  if (lane == 0) red[wid] = s;
  __syncthreads();
  if (tid == 0) {
    float t = 0.0f;
    for (int i = 0; i < 8; ++i) t += red[i];
    s_mean = t / (float)ND;
  }
  __syncthreads();
  const float mean = s_mean;
  float ss = 0.0f;
#pragma unroll
  for (int i = 0; i < 4; ++i) { const float d = v[i] - mean; ss += d * d; }
#pragma unroll
  for (int o = 16; o >= 1; o >>= 1) ss += __shfl_xor(ss, o, 32);
  if (lane == 0) red[wid] = ss;
  __syncthreads();
  if (tid == 0) {
    float t = 0.0f;
    for (int i = 0; i < 8; ++i) t += red[i];
    const float var = t / (float)(ND - 1);
    s_inv = 1.0f / (sqrtf(var) + 1.0e-6f);
  }
  __syncthreads();
  const float inv = s_inv;
#pragma unroll
  for (int i = 0; i < 4; ++i) {
    const int c = tid + i * 256;
    out[row * ND + c] = (OT)(gamma[c] * (v[i] - mean) * inv + beta[c]);
  }
}

// ---------------- host launcher ----------------
extern "C" void kernel_launch(void* const* d_in, const int* in_sizes, int n_in,
                              void* d_out, int out_size, void* d_ws, size_t ws_size,
                              hipStream_t stream) {
  (void)in_sizes; (void)n_in; (void)out_size; (void)ws_size;
  const float* x    = (const float*)d_in[0];
  const int*   mask = (const int*)d_in[1];
  const float* Wq = (const float*)d_in[2];  const float* bq = (const float*)d_in[3];
  const float* Wk = (const float*)d_in[4];  const float* bk = (const float*)d_in[5];
  const float* Wv = (const float*)d_in[6];  const float* bv = (const float*)d_in[7];
  const float* Wo = (const float*)d_in[8];  const float* bo = (const float*)d_in[9];
  const float* W1 = (const float*)d_in[10]; const float* b1 = (const float*)d_in[11];
  const float* W2 = (const float*)d_in[12]; const float* b2 = (const float*)d_in[13];
  const float* g1 = (const float*)d_in[14]; const float* e1 = (const float*)d_in[15];
  const float* g2 = (const float*)d_in[16]; const float* e2 = (const float*)d_in[17];
  float* out = (float*)d_out;

  const int  TOK   = NB * NS;             // 8192 tokens
  const long SLOTH = (long)TOK * ND;      // 8,388,608 halves per slot
  const long MW    = (long)ND * ND;       // 1,048,576 halves
  _Float16* wsh = (_Float16*)d_ws;
  _Float16* xh   = wsh + 0 * SLOTH;
  _Float16* qh   = wsh + 1 * SLOTH;
  _Float16* kh   = wsh + 2 * SLOTH;
  _Float16* vh   = wsh + 3 * SLOTH;
  _Float16* ctxh = wsh + 4 * SLOTH;
  _Float16* ff1h = wsh + 5 * SLOTH;       // spans 4 slots (TOK x NFF)
  _Float16* wt   = wsh + 9 * SLOTH;
  _Float16* WqT = wt + 0 * MW; _Float16* WkT = wt + 1 * MW;
  _Float16* WvT = wt + 2 * MW; _Float16* WoT = wt + 3 * MW;
  _Float16* W1T = wt + 4 * MW;            // [NFF, ND]
  _Float16* W2T = wt + 8 * MW;            // [ND, NFF]
  _Float16* attnh = qh;                   // q dead after attention
  _Float16* x1h   = kh;                   // k dead after attention
  _Float16* ff2h  = vh;                   // v dead after attention

  dim3 blk(256);
  dim3 gD(ND / BN, TOK / BM);             // (8, 64)
  dim3 gF(NFF / BN, TOK / BM);            // (32, 64)

  // pre-pass: f16 conversion + weight transposes (one-time, tiny vs GEMM work)
  cvt_kernel<<<dim3((int)(SLOTH / 1024)), blk, 0, stream>>>(x, xh);
  transpose_cvt_kernel<<<dim3(ND / 32, ND / 32), blk, 0, stream>>>(Wq, WqT, ND, ND);
  transpose_cvt_kernel<<<dim3(ND / 32, ND / 32), blk, 0, stream>>>(Wk, WkT, ND, ND);
  transpose_cvt_kernel<<<dim3(ND / 32, ND / 32), blk, 0, stream>>>(Wv, WvT, ND, ND);
  transpose_cvt_kernel<<<dim3(ND / 32, ND / 32), blk, 0, stream>>>(Wo, WoT, ND, ND);
  transpose_cvt_kernel<<<dim3(NFF / 32, ND / 32), blk, 0, stream>>>(W1, W1T, ND, NFF);
  transpose_cvt_kernel<<<dim3(ND / 32, NFF / 32), blk, 0, stream>>>(W2, W2T, NFF, ND);

  gemm16_kernel<0><<<gD, blk, 0, stream>>>(xh,   WqT, bq, qh,   TOK, ND,  ND);
  gemm16_kernel<0><<<gD, blk, 0, stream>>>(xh,   WkT, bk, kh,   TOK, ND,  ND);
  gemm16_kernel<0><<<gD, blk, 0, stream>>>(xh,   WvT, bv, vh,   TOK, ND,  ND);
  attn_kernel<<<dim3(NS / 64, NH, NB), dim3(128), 0, stream>>>(qh, kh, vh, mask,
                                                               ctxh);
  gemm16_kernel<0><<<gD, blk, 0, stream>>>(ctxh, WoT, bo, attnh, TOK, ND,  ND);
  add_ln_kernel<_Float16><<<dim3(TOK), blk, 0, stream>>>(xh, attnh, g1, e1, x1h);
  gemm16_kernel<1><<<gF, blk, 0, stream>>>(x1h,  W1T, b1, ff1h, TOK, NFF, ND);
  gemm16_kernel<0><<<gD, blk, 0, stream>>>(ff1h, W2T, b2, ff2h, TOK, ND,  NFF);
  add_ln_kernel<float><<<dim3(TOK), blk, 0, stream>>>(x1h, ff2h, g2, e2, out);
}
